// MultiHeadAttention_28724741275823
// MI455X (gfx1250) — compile-verified, matrix operand order checked by reading, not run on hardware
//
#include <hip/hip_runtime.h>
#include <hip/hip_bf16.h>

// MHA forward, B=2,S=2048,E=1024,H=16,D=64, gfx1250 wave32 WMMA.
// proj(q,wq)->qh[B,H,S,D]f16 ; proj(k,wq)->kh (source bug: k uses wq)
// proj(v,wv)->vt TRANSPOSED [B,H,D,S]f16
// fused attention: 3-sweep (rowmax / rowsum / p+AV), writes attn exactly once
// out = ctx @ wo + wo_b
// Roofline: attn (512MB) is the only HBM stream (L2=192MB holds the rest);
// fused kernel keeps attn traffic at 0.5GB (vs 2GB unfused).

#define B_ 2
#define S_ 2048
#define E_ 1024
#define H_ 16
#define D_ 64

typedef __attribute__((ext_vector_type(16))) _Float16 v16h;
typedef __attribute__((ext_vector_type(8)))  _Float16 v8h;
typedef __attribute__((ext_vector_type(2)))  _Float16 v2h;
typedef __attribute__((ext_vector_type(8)))  float    v8f;

__device__ __forceinline__ v8f zero8() {
  v8f z;
#pragma unroll
  for (int i = 0; i < 8; ++i) z[i] = 0.0f;
  return z;
}

// 16x32 f16 fragment per ISA 7.12.2: lane l -> row/col l%16,
// elems 0..7 = K[8h..8h+7], elems 8..15 = K[16+8h..], h=l/16. 2x ds_load_b128.
__device__ __forceinline__ v16h make_frag(const _Float16* lo_p, const _Float16* hi_p) {
  v8h lo = *(const v8h*)lo_p;
  v8h hi = *(const v8h*)hi_p;
  v16h r;
#pragma unroll
  for (int i = 0; i < 8; ++i) { r[i] = lo[i]; r[i + 8] = hi[i]; }
  return r;
}

__device__ __forceinline__ v8f wmma16x16x32(v16h a, v16h b, v8f c) {
  return __builtin_amdgcn_wmma_f32_16x16x32_f16(false, a, false, b, (short)0, c,
                                                false, false);
}

// Butterfly reductions over the 16-lane halves (C-fragment col groups).
__device__ __forceinline__ float redmax16(float v) {
#pragma unroll
  for (int off = 1; off < 16; off <<= 1) v = fmaxf(v, __shfl_xor(v, off, 32));
  return v;
}
__device__ __forceinline__ float redsum16(float v) {
#pragma unroll
  for (int off = 1; off < 16; off <<= 1) v += __shfl_xor(v, off, 32);
  return v;
}

// ---------------------------------------------------------------------------
// Kernel 1: C[4096,1024] = A(f32) @ W(f32) + bias -> f16 split-head.
// VTRANS=0: out[((b*H+h)*S+s)*D+d]   (qh/kh layout)
// VTRANS=1: out[((b*H+h)*D+d)*S+s]   (vt layout for fused AV)
// Block tile 128x128, 8 waves (4m x 2n), wave tile 32x64, BK=32.
// ---------------------------------------------------------------------------
template <int VTRANS>
__global__ __launch_bounds__(256) void proj_split_kernel(
    const float* __restrict__ A, const float* __restrict__ W,
    const float* __restrict__ bias, _Float16* __restrict__ out) {
  __shared__ _Float16 sA[128 * 40];
  __shared__ _Float16 sB[128 * 40];
  const int tid  = threadIdx.x;
  const int lane = tid & 31;
  const int wave = tid >> 5;
  const int wm   = wave >> 1;
  const int wn   = wave & 1;
  const int lm   = lane & 15;
  const int lh   = lane >> 4;
  const int m0   = blockIdx.y * 128;
  const int n0   = blockIdx.x * 128;

  v8f acc[2][4];
#pragma unroll
  for (int i = 0; i < 2; ++i)
#pragma unroll
    for (int j = 0; j < 4; ++j) acc[i][j] = zero8();

  for (int k0 = 0; k0 < E_; k0 += 32) {
#pragma unroll
    for (int i = 0; i < 8; ++i) {          // A tile 128x32: float2 -> packed b32
      int idx = tid + 256 * i;
      int r = idx >> 4, c2 = idx & 15;
      float2 f2 = *(const float2*)&A[(size_t)(m0 + r) * E_ + k0 + c2 * 2];
      v2h p;
      p.x = (_Float16)f2.x;
      p.y = (_Float16)f2.y;
      *(v2h*)&sA[r * 40 + c2 * 2] = p;
    }
#pragma unroll
    for (int i = 0; i < 16; ++i) {         // W tile 32x128 -> [n][k] (transpose)
      int idx = tid + 256 * i;
      int kl = idx >> 7, c = idx & 127;
      sB[c * 40 + kl] = (_Float16)W[(size_t)(k0 + kl) * E_ + n0 + c];
    }
    __syncthreads();
    v16h af[2], bf[4];
#pragma unroll
    for (int i = 0; i < 2; ++i) {
      int m = wm * 32 + i * 16 + lm;
      af[i] = make_frag(&sA[m * 40 + 8 * lh], &sA[m * 40 + 16 + 8 * lh]);
    }
#pragma unroll
    for (int j = 0; j < 4; ++j) {
      int n = wn * 64 + j * 16 + lm;
      bf[j] = make_frag(&sB[n * 40 + 8 * lh], &sB[n * 40 + 16 + 8 * lh]);
    }
#pragma unroll
    for (int i = 0; i < 2; ++i)
#pragma unroll
      for (int j = 0; j < 4; ++j) acc[i][j] = wmma16x16x32(af[i], bf[j], acc[i][j]);
    __syncthreads();
  }

#pragma unroll
  for (int i = 0; i < 2; ++i) {
    int gm0 = m0 + wm * 32 + i * 16 + lh * 8;
#pragma unroll
    for (int j = 0; j < 4; ++j) {
      int n = n0 + wn * 64 + j * 16 + lm;
      float bv = bias[n];
      int h = n >> 6, d = n & 63;
#pragma unroll
      for (int r = 0; r < 8; ++r) {
        int m = gm0 + r;
        int b = m >> 11, s = m & (S_ - 1);
        _Float16 hv = (_Float16)(acc[i][j][r] + bv);
        if (VTRANS)
          out[((size_t)(b * H_ + h) * D_ + d) * S_ + s] = hv;
        else
          out[(((size_t)(b * H_ + h) * S_) + s) * D_ + d] = hv;
      }
    }
  }
}

// ---------------------------------------------------------------------------
// Kernel 2: fused attention. One workgroup = 128 query rows of one (b,h).
// Wave tile 16q; 8 waves. Sweeps: 1) rowmax 2) rowsum 3) p -> attn + AV.
// attn is written exactly once (the only large HBM stream in the pipeline).
// ---------------------------------------------------------------------------
__global__ __launch_bounds__(256) void fused_attn_kernel(
    const _Float16* __restrict__ qh, const _Float16* __restrict__ kh,
    const _Float16* __restrict__ vt, const float* __restrict__ mask,
    float* __restrict__ attn, _Float16* __restrict__ ctx) {
  __shared__ _Float16 sQ[128 * 72];  // [q_local][d]
  __shared__ _Float16 sK[64 * 72];   // [key_local][d]
  __shared__ _Float16 sV[64 * 72];   // [d][key_local] (vt rows copy straight in)
  __shared__ _Float16 sP[128 * 72];  // [q_local][key_local] f16 probs
  const int tid  = threadIdx.x;
  const int lane = tid & 31;
  const int wave = tid >> 5;
  const int lm   = lane & 15;
  const int lh   = lane >> 4;
  const int bh   = blockIdx.y;
  const int b    = bh >> 4, h = bh & 15;
  const int m0   = blockIdx.x * 128;
  const _Float16* Q  = qh + (size_t)bh * S_ * D_;
  const _Float16* Kp = kh + (size_t)bh * S_ * D_;
  const _Float16* Vt = vt + (size_t)bh * D_ * S_;
  const float* mrow  = mask + b * S_;
  float* attn_bh     = attn + (size_t)bh * S_ * S_;

  // Stage qh block [128][64] once.
#pragma unroll
  for (int i = 0; i < 4; ++i) {
    int idx = tid + 256 * i;
    int r = idx >> 3, c = idx & 7;
    *(uint4*)&sQ[r * 72 + c * 8] = *(const uint4*)&Q[(size_t)(m0 + r) * D_ + c * 8];
  }
  __syncthreads();
  v16h aq[2];  // persistent A-fragments: rows wave*16.., K=64 in two halves
  {
    int m = wave * 16 + lm;
    aq[0] = make_frag(&sQ[m * 72 + 8 * lh], &sQ[m * 72 + 16 + 8 * lh]);
    aq[1] = make_frag(&sQ[m * 72 + 32 + 8 * lh], &sQ[m * 72 + 48 + 8 * lh]);
  }

  // ---- Sweep 1: row max -------------------------------------------------
  float rm[8];
#pragma unroll
  for (int r = 0; r < 8; ++r) rm[r] = -3.0e38f;

  for (int k0 = 0; k0 < S_; k0 += 64) {
#pragma unroll
    for (int i = 0; i < 2; ++i) {
      int idx = tid + 256 * i;
      int r = idx >> 3, c = idx & 7;
      *(uint4*)&sK[r * 72 + c * 8] = *(const uint4*)&Kp[(size_t)(k0 + r) * D_ + c * 8];
    }
    __syncthreads();
    v8f lacc[4];
#pragma unroll
    for (int j = 0; j < 4; ++j) lacc[j] = zero8();
#pragma unroll
    for (int kk = 0; kk < 2; ++kk)
#pragma unroll
      for (int j = 0; j < 4; ++j) {
        int n = j * 16 + lm;
        v16h bf = make_frag(&sK[n * 72 + kk * 32 + 8 * lh],
                            &sK[n * 72 + kk * 32 + 16 + 8 * lh]);
        lacc[j] = wmma16x16x32(aq[kk], bf, lacc[j]);
      }
    float mk[4];
#pragma unroll
    for (int j = 0; j < 4; ++j) mk[j] = mrow[k0 + j * 16 + lm] * -1e9f;
#pragma unroll
    for (int r = 0; r < 8; ++r) {
      float v = -3.0e38f;
#pragma unroll
      for (int j = 0; j < 4; ++j) v = fmaxf(v, lacc[j][r] * 0.125f + mk[j]);
      rm[r] = fmaxf(rm[r], redmax16(v));
    }
    __syncthreads();
  }

  // ---- Sweep 2: row sum of exp ------------------------------------------
  float rs[8];
#pragma unroll
  for (int r = 0; r < 8; ++r) rs[r] = 0.0f;

  for (int k0 = 0; k0 < S_; k0 += 64) {
#pragma unroll
    for (int i = 0; i < 2; ++i) {
      int idx = tid + 256 * i;
      int r = idx >> 3, c = idx & 7;
      *(uint4*)&sK[r * 72 + c * 8] = *(const uint4*)&Kp[(size_t)(k0 + r) * D_ + c * 8];
    }
    __syncthreads();
    v8f lacc[4];
#pragma unroll
    for (int j = 0; j < 4; ++j) lacc[j] = zero8();
#pragma unroll
    for (int kk = 0; kk < 2; ++kk)
#pragma unroll
      for (int j = 0; j < 4; ++j) {
        int n = j * 16 + lm;
        v16h bf = make_frag(&sK[n * 72 + kk * 32 + 8 * lh],
                            &sK[n * 72 + kk * 32 + 16 + 8 * lh]);
        lacc[j] = wmma16x16x32(aq[kk], bf, lacc[j]);
      }
    float mk[4];
#pragma unroll
    for (int j = 0; j < 4; ++j) mk[j] = mrow[k0 + j * 16 + lm] * -1e9f;
#pragma unroll
    for (int r = 0; r < 8; ++r) {
      float e = 0.0f;
#pragma unroll
      for (int j = 0; j < 4; ++j)
        e += __expf(lacc[j][r] * 0.125f + mk[j] - rm[r]);
      rs[r] += redsum16(e);
    }
    __syncthreads();
  }

  float rinv[8];
#pragma unroll
  for (int r = 0; r < 8; ++r) rinv[r] = 1.0f / rs[r];

  // ---- Sweep 3: p = exp(l-max)/sum -> attn (only write) + AV GEMM -------
  v8f cacc[4];
#pragma unroll
  for (int j = 0; j < 4; ++j) cacc[j] = zero8();

  for (int k0 = 0; k0 < S_; k0 += 64) {
#pragma unroll
    for (int i = 0; i < 2; ++i) {
      int idx = tid + 256 * i;
      int r = idx >> 3, c = idx & 7;
      *(uint4*)&sK[r * 72 + c * 8] = *(const uint4*)&Kp[(size_t)(k0 + r) * D_ + c * 8];
      // vt rows: sV[d][key] straight copy, no transpose needed
      *(uint4*)&sV[r * 72 + c * 8] = *(const uint4*)&Vt[(size_t)r * S_ + k0 + c * 8];
    }
    __syncthreads();
    v8f lacc[4];
#pragma unroll
    for (int j = 0; j < 4; ++j) lacc[j] = zero8();
#pragma unroll
    for (int kk = 0; kk < 2; ++kk)
#pragma unroll
      for (int j = 0; j < 4; ++j) {
        int n = j * 16 + lm;
        v16h bf = make_frag(&sK[n * 72 + kk * 32 + 8 * lh],
                            &sK[n * 72 + kk * 32 + 16 + 8 * lh]);
        lacc[j] = wmma16x16x32(aq[kk], bf, lacc[j]);
      }
    float mk[4];
#pragma unroll
    for (int j = 0; j < 4; ++j) mk[j] = mrow[k0 + j * 16 + lm] * -1e9f;
#pragma unroll
    for (int j = 0; j < 4; ++j) {
      int colg = k0 + j * 16 + lm;
#pragma unroll
      for (int r = 0; r < 8; ++r) {
        float p = __expf(lacc[j][r] * 0.125f + mk[j] - rm[r]) * rinv[r];
        int mloc = wave * 16 + lh * 8 + r;
        attn_bh[(size_t)(m0 + mloc) * S_ + colg] = p;       // sole attn write
        sP[mloc * 72 + j * 16 + lm] = (_Float16)p;          // C-layout -> A-layout
      }
    }
    // AV: rows are wave-private in sP, so only DScnt ordering (compiler) needed.
    v16h af[2];
    {
      int m = wave * 16 + lm;
      af[0] = make_frag(&sP[m * 72 + 8 * lh], &sP[m * 72 + 16 + 8 * lh]);
      af[1] = make_frag(&sP[m * 72 + 32 + 8 * lh], &sP[m * 72 + 48 + 8 * lh]);
    }
#pragma unroll
    for (int kk = 0; kk < 2; ++kk)
#pragma unroll
      for (int j = 0; j < 4; ++j) {
        int n = j * 16 + lm;
        v16h bf = make_frag(&sV[n * 72 + kk * 32 + 8 * lh],
                            &sV[n * 72 + kk * 32 + 16 + 8 * lh]);
        cacc[j] = wmma16x16x32(af[kk], bf, cacc[j]);
      }
    __syncthreads();
  }

  // ctx in concat layout [B,S,E] f16
#pragma unroll
  for (int j = 0; j < 4; ++j) {
    int n = j * 16 + lm;  // d
#pragma unroll
    for (int r = 0; r < 8; ++r) {
      int m = m0 + wave * 16 + lh * 8 + r;  // q
      ctx[((size_t)(b * S_ + m)) * E_ + h * 64 + n] = (_Float16)cacc[j][r];
    }
  }
}

// ---------------------------------------------------------------------------
// Kernel 3: out[4096,1024](f32) = ctx(f16) @ wo(f32) + wo_b.
// ---------------------------------------------------------------------------
__global__ __launch_bounds__(256) void outproj_kernel(
    const _Float16* __restrict__ A, const float* __restrict__ W,
    const float* __restrict__ bias, float* __restrict__ out) {
  __shared__ _Float16 sA[128 * 40];
  __shared__ _Float16 sB[128 * 40];
  const int tid  = threadIdx.x;
  const int lane = tid & 31;
  const int wave = tid >> 5;
  const int wm   = wave >> 1;
  const int wn   = wave & 1;
  const int lm   = lane & 15;
  const int lh   = lane >> 4;
  const int m0   = blockIdx.y * 128;
  const int n0   = blockIdx.x * 128;

  v8f acc[2][4];
#pragma unroll
  for (int i = 0; i < 2; ++i)
#pragma unroll
    for (int j = 0; j < 4; ++j) acc[i][j] = zero8();

  for (int k0 = 0; k0 < E_; k0 += 32) {
#pragma unroll
    for (int i = 0; i < 8; ++i) {          // A tile 128x32 f16, b32 copies
      int idx = tid + 256 * i;
      int r = idx >> 4, c2 = idx & 15;
      *(unsigned int*)&sA[r * 40 + c2 * 2] =
          *(const unsigned int*)&A[(size_t)(m0 + r) * E_ + k0 + c2 * 2];
    }
#pragma unroll
    for (int i = 0; i < 16; ++i) {         // W tile 32x128 -> [n][k]
      int idx = tid + 256 * i;
      int kl = idx >> 7, c = idx & 127;
      sB[c * 40 + kl] = (_Float16)W[(size_t)(k0 + kl) * E_ + n0 + c];
    }
    __syncthreads();
    v16h af[2], bf[4];
#pragma unroll
    for (int i = 0; i < 2; ++i) {
      int m = wm * 32 + i * 16 + lm;
      af[i] = make_frag(&sA[m * 40 + 8 * lh], &sA[m * 40 + 16 + 8 * lh]);
    }
#pragma unroll
    for (int j = 0; j < 4; ++j) {
      int n = wn * 64 + j * 16 + lm;
      bf[j] = make_frag(&sB[n * 40 + 8 * lh], &sB[n * 40 + 16 + 8 * lh]);
    }
#pragma unroll
    for (int i = 0; i < 2; ++i)
#pragma unroll
      for (int j = 0; j < 4; ++j) acc[i][j] = wmma16x16x32(af[i], bf[j], acc[i][j]);
    __syncthreads();
  }

#pragma unroll
  for (int i = 0; i < 2; ++i) {
    int gm0 = m0 + wm * 32 + i * 16 + lh * 8;
#pragma unroll
    for (int j = 0; j < 4; ++j) {
      int n = n0 + wn * 64 + j * 16 + lm;
      float bv = bias[n];
#pragma unroll
      for (int r = 0; r < 8; ++r) {
        int m = gm0 + r;
        out[(size_t)m * E_ + n] = acc[i][j][r] + bv;
      }
    }
  }
}

extern "C" void kernel_launch(void* const* d_in, const int* in_sizes, int n_in,
                              void* d_out, int out_size, void* d_ws, size_t ws_size,
                              hipStream_t stream) {
  const float* v    = (const float*)d_in[0];
  const float* k    = (const float*)d_in[1];
  const float* q    = (const float*)d_in[2];
  const float* mask = (const float*)d_in[3];
  const float* wq_w = (const float*)d_in[4];
  const float* wq_b = (const float*)d_in[5];
  const float* wv_w = (const float*)d_in[6];
  const float* wv_b = (const float*)d_in[7];
  const float* wo_w = (const float*)d_in[8];
  const float* wo_b = (const float*)d_in[9];

  float* out  = (float*)d_out;                        // [B,S,E]
  float* attn = out + (size_t)B_ * S_ * E_;           // [B,H,S,S]

  _Float16* qh  = (_Float16*)d_ws;                    // [B,H,S,D] f16
  _Float16* kh  = qh + (size_t)B_ * H_ * S_ * D_;     // [B,H,S,D] f16
  _Float16* vt  = kh + (size_t)B_ * H_ * S_ * D_;     // [B,H,D,S] f16 (transposed)
  _Float16* ctx = vt + (size_t)B_ * H_ * S_ * D_;     // [B,S,E] f16

  dim3 blk(256);
  dim3 gproj(E_ / 128, (B_ * S_) / 128);              // (8, 32)
  proj_split_kernel<0><<<gproj, blk, 0, stream>>>(q, wq_w, wq_b, qh);
  proj_split_kernel<0><<<gproj, blk, 0, stream>>>(k, wq_w, wq_b, kh);  // k uses wq (source bug)
  proj_split_kernel<1><<<gproj, blk, 0, stream>>>(v, wv_w, wv_b, vt);

  fused_attn_kernel<<<dim3(S_ / 128, B_ * H_), blk, 0, stream>>>(
      qh, kh, vt, mask, attn, ctx);

  outproj_kernel<<<gproj, blk, 0, stream>>>(ctx, wo_w, wo_b, out);
}
